// Attend_72430328479879
// MI455X (gfx1250) — matrix operand order repaired and checked by reading, not compiled, not run on hardware
//
#include <hip/hip_runtime.h>
#include <hip/hip_bf16.h>

// ---------------------------------------------------------------------------
// Decomposable-attention "Attend" block on MI455X (gfx1250).
// - every GEMM: v_wmma_f32_16x16x32_bf16, fp32 accumulate
// - LDS double-buffered, staged with GLOBAL_LOAD_ASYNC_TO_LDS_B128 (ASYNCcnt)
//   when the toolchain exposes the builtin (sync fallback otherwise)
// - all GEMMs are A(MxK) @ B(NxK)^T so staging is pure K-contiguous copies
// ---------------------------------------------------------------------------

#if __has_builtin(__builtin_amdgcn_global_load_async_to_lds_b128) && \
    __has_builtin(__builtin_amdgcn_s_wait_asynccnt)
#define USE_ASYNC 1
#else
#define USE_ASYNC 0
#endif

typedef __attribute__((ext_vector_type(16))) __bf16 v16bf;
typedef __attribute__((ext_vector_type(8)))  float  v8f;
typedef __attribute__((ext_vector_type(4)))  int    v4i;   // builtin's pointee type

__device__ __forceinline__ unsigned short f2bf(float f) {
    unsigned u = __builtin_bit_cast(unsigned, f);
    unsigned r = u + 0x7FFFu + ((u >> 16) & 1u);   // round-to-nearest-even
    return (unsigned short)(r >> 16);
}

// copy 32 bytes (16 bf16) global -> LDS, per lane
__device__ __forceinline__ void cp32(const unsigned short* __restrict__ g,
                                     unsigned short* l) {
#if USE_ASYNC
    __builtin_amdgcn_global_load_async_to_lds_b128((v4i*)g,       (v4i*)l,       0, 0);
    __builtin_amdgcn_global_load_async_to_lds_b128((v4i*)(g + 8), (v4i*)(l + 8), 0, 0);
#else
    uint4 p0 = *(const uint4*)g;
    uint4 p1 = *(const uint4*)(g + 8);
    *(uint4*)l       = p0;
    *(uint4*)(l + 8) = p1;
#endif
}

template <int N>
__device__ __forceinline__ void wait_async() {
#if USE_ASYNC
    __builtin_amdgcn_s_wait_asynccnt(N);
#endif
}

// ------------------------------ fp32 -> bf16 --------------------------------
__global__ __launch_bounds__(256)
void cvt_f32_bf16(const float* __restrict__ in, unsigned short* __restrict__ out,
                  long long n) {
    long long i = (long long)blockIdx.x * blockDim.x + threadIdx.x;
    long long stride = (long long)gridDim.x * blockDim.x;
    for (; i < n; i += stride) out[i] = f2bf(in[i]);
}

// --------------------- fp32 (R,C) -> bf16 transposed (C,R) ------------------
__global__ __launch_bounds__(256)
void cvt_t_bf16(const float* __restrict__ in, unsigned short* __restrict__ out,
                int R, int C, long long ibatch, long long obatch) {
    __shared__ float t[32][33];
    const int tx = threadIdx.x & 31;
    const int ty = threadIdx.x >> 5;            // 0..7
    const int c0 = blockIdx.x * 32;
    const int r0 = blockIdx.y * 32;
    const long long b = blockIdx.z;
    const float* src = in + b * ibatch;
    unsigned short* dst = out + b * obatch;
#pragma unroll
    for (int j = 0; j < 4; ++j)
        t[ty + j * 8][tx] = src[(long long)(r0 + ty + j * 8) * C + c0 + tx];
    __syncthreads();
#pragma unroll
    for (int j = 0; j < 4; ++j)
        dst[(long long)(c0 + ty + j * 8) * R + r0 + tx] = f2bf(t[tx][ty + j * 8]);
}

// ------------------------------ WMMA GEMM (BT) ------------------------------
// C = act(A @ B^T [+ bias]).  A: MxK bf16 row-major.  B: NxK bf16 row-major.
// FUSED: +bias[col], ReLU, bf16 out.  else: fp32 out.
// Block tile 128x128, 8 waves, wave tile 32x64 (2x4 WMMA tiles).
// Double-buffered LDS; async global->LDS staging when available.
#define LDSP 40   // padded LDS row stride (elements); 80B = 5*16B (b128-aligned)

template <bool FUSED>
__global__ __launch_bounds__(256)
void gemm_bt_wmma(const unsigned short* __restrict__ A,
                  const unsigned short* __restrict__ Bm,
                  const float* __restrict__ bias,
                  void* __restrict__ Cout,
                  int M, int N, int K,
                  long long aBatch, long long bBatch, long long cBatch) {
    __shared__ unsigned short lds_a[2][128][LDSP];   // [buf][m][k]
    __shared__ unsigned short lds_b[2][128][LDSP];   // [buf][n][k]

    const int tid  = threadIdx.x;
    const int lane = tid & 31;
    const int wave = tid >> 5;
    const int wm   = wave & 3;     // 4 waves along M
    const int wn   = wave >> 2;    // 2 waves along N
    const int ln15 = lane & 15;
    const int half = lane >> 4;    // lane half per ISA 7.12.2 layouts

    const int n0 = blockIdx.x * 128;
    const int m0 = blockIdx.y * 128;
    const long long z = blockIdx.z;

    const unsigned short* Ab = A  + z * aBatch;
    const unsigned short* Bb = Bm + z * bBatch;

    // staging assignment: 2 threads per row, 16 elements (32B) each
    const int row_s = tid >> 1;
    const int cb_s  = (tid & 1) * 16;

    v8f acc[2][4];
    const v8f vzero = {0.f, 0.f, 0.f, 0.f, 0.f, 0.f, 0.f, 0.f};
#pragma unroll
    for (int m = 0; m < 2; ++m)
#pragma unroll
        for (int n = 0; n < 4; ++n) acc[m][n] = vzero;

    union Frag { v16bf v; uint4 q[2]; };

    auto stage = [&](int k0, int b) {
        cp32(Ab + (long long)(m0 + row_s) * K + k0 + cb_s, &lds_a[b][row_s][cb_s]);
        cp32(Bb + (long long)(n0 + row_s) * K + k0 + cb_s, &lds_b[b][row_s][cb_s]);
    };

    const int nch = K >> 5;       // K / 32
    int buf = 0;
    stage(0, 0);                  // prologue: chunk 0 -> buffer 0
    for (int ci = 0; ci < nch; ++ci) {
        if (ci + 1 < nch) {
            stage((ci + 1) << 5, buf ^ 1);  // prefetch next chunk
            wait_async<4>();                // previous 4 loads landed
        } else {
            wait_async<0>();
        }
        __syncthreads();

        // ---- fragments (2x b128 each, per ISA bf16 VGPR layouts) ----
        Frag afr[2], bfr[4];
#pragma unroll
        for (int m = 0; m < 2; ++m) {
            const int row = wm * 32 + m * 16 + ln15;
            afr[m].q[0] = *(const uint4*)&lds_a[buf][row][half * 8];
            afr[m].q[1] = *(const uint4*)&lds_a[buf][row][16 + half * 8];
        }
#pragma unroll
        for (int n = 0; n < 4; ++n) {
            const int col = wn * 64 + n * 16 + ln15;
            bfr[n].q[0] = *(const uint4*)&lds_b[buf][col][half * 8];
            bfr[n].q[1] = *(const uint4*)&lds_b[buf][col][16 + half * 8];
        }
        // ---- 8 WMMAs per K-chunk per wave ----
#pragma unroll
        for (int m = 0; m < 2; ++m)
#pragma unroll
            for (int n = 0; n < 4; ++n)
                acc[m][n] = __builtin_amdgcn_wmma_f32_16x16x32_bf16(
                    false, afr[m].v, false, bfr[n].v,
                    (short)0, acc[m][n], false, false);
        __syncthreads();          // all reads of buf done before its reuse
        buf ^= 1;
    }

    // ---- epilogue ----
#pragma unroll
    for (int m = 0; m < 2; ++m) {
#pragma unroll
        for (int n = 0; n < 4; ++n) {
            const int gcol = n0 + wn * 64 + n * 16 + ln15;
            const long long rowbase = m0 + wm * 32 + m * 16 + half * 8;
#pragma unroll
            for (int r = 0; r < 8; ++r) {
                float v = acc[m][n][r];
                const long long idx = (rowbase + r) * (long long)N + gcol;
                if (FUSED) {
                    v += bias[gcol];
                    v = v > 0.f ? v : 0.f;
                    ((unsigned short*)Cout)[cBatch * z + idx] = f2bf(v);
                } else {
                    ((float*)Cout)[cBatch * z + idx] = v;
                }
            }
        }
    }
}

// ------------------------------ softmax -------------------------------------
// One block per (row, batch).  Row length fixed at 1024 (4 elems/thread).
// estride lets the same kernel do row-softmax of e and of e^T.
__global__ __launch_bounds__(256)
void softmax_bf16(const float* __restrict__ e, unsigned short* __restrict__ P,
                  long long bstrideE, long long rstride, long long estride,
                  long long bstrideP) {
    __shared__ float red[256];
    const int tid = threadIdx.x;
    const long long row = blockIdx.x;
    const long long b   = blockIdx.y;
    const float* src = e + b * bstrideE + row * rstride;

    float v[4];
    float mx = -3.402823466e38f;
#pragma unroll
    for (int i = 0; i < 4; ++i) {
        v[i] = src[(long long)(tid + i * 256) * estride];
        mx = fmaxf(mx, v[i]);
    }
    red[tid] = mx; __syncthreads();
    for (int s = 128; s > 0; s >>= 1) {
        if (tid < s) red[tid] = fmaxf(red[tid], red[tid + s]);
        __syncthreads();
    }
    mx = red[0]; __syncthreads();

    float sum = 0.f;
#pragma unroll
    for (int i = 0; i < 4; ++i) { v[i] = __expf(v[i] - mx); sum += v[i]; }
    red[tid] = sum; __syncthreads();
    for (int s = 128; s > 0; s >>= 1) {
        if (tid < s) red[tid] += red[tid + s];
        __syncthreads();
    }
    const float inv = 1.0f / red[0];

    unsigned short* dst = P + b * bstrideP + row * 1024;
#pragma unroll
    for (int i = 0; i < 4; ++i) dst[tid + i * 256] = f2bf(v[i] * inv);
}

// ------------------------------ launcher ------------------------------------
extern "C" void kernel_launch(void* const* d_in, const int* in_sizes, int n_in,
                              void* d_out, int out_size, void* d_ws, size_t ws_size,
                              hipStream_t stream) {
    constexpr int  BSZ = 16, TA = 1024, TB = 1024, EMB = 768, HID = 1024;
    constexpr long long ll_AE = (long long)BSZ * TA * EMB;   // elems of A
    constexpr long long ll_BE = (long long)BSZ * TB * EMB;
    constexpr long long ll_AH = (long long)BSZ * TA * HID;   // elems of f_A
    constexpr long long ll_BH = (long long)BSZ * TB * HID;

    const float* A  = (const float*)d_in[0];
    const float* Bx = (const float*)d_in[1];
    const float* W1 = (const float*)d_in[2];
    const float* b1 = (const float*)d_in[3];
    const float* W2 = (const float*)d_in[4];
    const float* b2 = (const float*)d_in[5];
    float* out = (float*)d_out;

    // ---- workspace layout (256B aligned) ----
    char* ws = (char*)d_ws;
    size_t off = 0;
    auto alloc = [&](size_t bytes) {
        char* p = ws + off;
        off += (bytes + 255) & ~(size_t)255;
        return p;
    };
    unsigned short* Xbf  = (unsigned short*)alloc((ll_AE + ll_BE) * 2); // [Abf|Bbf] row-major
    unsigned short* Abf  = Xbf;
    unsigned short* Bbf  = Xbf + ll_AE;
    unsigned short* W1T  = (unsigned short*)alloc((size_t)HID * EMB * 2); // (1024,768)
    unsigned short* W2T  = (unsigned short*)alloc((size_t)HID * HID * 2); // (1024,1024)
    unsigned short* AbfT = (unsigned short*)alloc(ll_AE * 2);             // (16,768,1024)
    unsigned short* BbfT = (unsigned short*)alloc(ll_BE * 2);             // (16,768,1024)
    unsigned short* h1bf = (unsigned short*)alloc((ll_AH + ll_BH) * 2);   // reused: P1|P2
    unsigned short* Fbf  = (unsigned short*)alloc((ll_AH + ll_BH) * 2);   // [fA|fB]
    unsigned short* fAbf = Fbf;
    unsigned short* fBbf = Fbf + ll_AH;
    float*          e    = (float*)alloc((size_t)BSZ * TA * TB * 4);
    unsigned short* P1   = h1bf;                          // (16,1024,1024) bf16
    unsigned short* P2   = h1bf + (long long)BSZ * TA * TB;

    dim3 blk(256);

    // ---- 1. conversions (row-major MLP inputs) ----
    cvt_f32_bf16<<<2048, blk, 0, stream>>>(A,  Abf, ll_AE);
    cvt_f32_bf16<<<2048, blk, 0, stream>>>(Bx, Bbf, ll_BE);

    // ---- 2. transpose-convert operands used on the B side of GEMMs ----
    cvt_t_bf16<<<dim3(HID / 32, EMB / 32, 1),  blk, 0, stream>>>(
        W1, W1T, EMB, HID, 0, 0);
    cvt_t_bf16<<<dim3(HID / 32, HID / 32, 1),  blk, 0, stream>>>(
        W2, W2T, HID, HID, 0, 0);
    cvt_t_bf16<<<dim3(EMB / 32, TA / 32, BSZ), blk, 0, stream>>>(
        A,  AbfT, TA, EMB, (long long)TA * EMB, (long long)TA * EMB);
    cvt_t_bf16<<<dim3(EMB / 32, TB / 32, BSZ), blk, 0, stream>>>(
        Bx, BbfT, TB, EMB, (long long)TB * EMB, (long long)TB * EMB);

    // ---- 3. MLP: h1 = relu(X @ W1 + b1)   (X = [A;B] -> 32768 x 768) ----
    dim3 g1(HID / 128, (2 * BSZ * TA) / 128, 1);
    gemm_bt_wmma<true><<<g1, blk, 0, stream>>>(
        Xbf, W1T, b1, h1bf, 2 * BSZ * TA, HID, EMB, 0, 0, 0);

    // ---- 4. MLP: F = relu(h1 @ W2 + b2)  -> [f_A | f_B] bf16 ----
    gemm_bt_wmma<true><<<g1, blk, 0, stream>>>(
        h1bf, W2T, b2, Fbf, 2 * BSZ * TA, HID, HID, 0, 0, 0);

    // ---- 5. e = f_A @ f_B^T  (batched, fp32) ----
    dim3 g2(TB / 128, TA / 128, BSZ);
    gemm_bt_wmma<false><<<g2, blk, 0, stream>>>(
        fAbf, fBbf, nullptr, e, TA, TB, HID,
        (long long)TA * HID, (long long)TB * HID, (long long)TA * TB);

    // ---- 6. P1 = softmax_rows(e), P2 = softmax_rows(e^T)  (bf16) ----
    softmax_bf16<<<dim3(TA, BSZ), blk, 0, stream>>>(
        e, P1, (long long)TA * TB, TB, 1, (long long)TA * TB);
    softmax_bf16<<<dim3(TB, BSZ), blk, 0, stream>>>(
        e, P2, (long long)TA * TB, 1, TB, (long long)TA * TB);

    // ---- 7. beta = P1 @ B = P1 @ BbfT^T -> out[0 : 12M) fp32 ----
    dim3 g3(EMB / 128, TA / 128, BSZ);
    gemm_bt_wmma<false><<<g3, blk, 0, stream>>>(
        P1, BbfT, nullptr, out, TA, EMB, TB,
        (long long)TA * TB, (long long)TB * EMB, (long long)TA * EMB);

    // ---- 8. alpha = P2 @ A = P2 @ AbfT^T -> out[12M : 24M) fp32 ----
    gemm_bt_wmma<false><<<g3, blk, 0, stream>>>(
        P2, AbfT, nullptr, out + ll_AE, TB, EMB, TA,
        (long long)TB * TA, (long long)TA * EMB, (long long)TB * EMB);
}
